// TaskAlignedAssigner_65919158059321
// MI455X (gfx1250) — matrix operand order repaired and checked by reading, not compiled, not run on hardware
//
#include <hip/hip_runtime.h>
#include <hip/hip_fp16.h>

#define BS    16
#define A_    8400
#define C_    80
#define NGT   60
#define TOPK_ 13
#define EPSF  1e-9f
#define TPB   256

typedef __attribute__((ext_vector_type(16))) _Float16 v16h;
typedef __attribute__((ext_vector_type(8)))  float    v8f;

__device__ __forceinline__ float iou_box(float gx0, float gy0, float gx1, float gy1, float garea,
                                         float px0, float py0, float px1, float py1) {
  float ltx = fmaxf(gx0, px0), lty = fmaxf(gy0, py0);
  float rbx = fminf(gx1, px1), rby = fminf(gy1, py1);
  float w = fmaxf(rbx - ltx, 0.0f), h = fmaxf(rby - lty, 0.0f);
  float ov = w * h;
  float parea = (px1 - px0) * (py1 - py0);
  return ov / (garea + parea - ov + EPSF);
}

// ---------------------------------------------------------------- K0: zero ws
__global__ void zero_kernel(unsigned int* __restrict__ cnt,
                            unsigned long long* __restrict__ best,
                            unsigned int* __restrict__ posA,
                            unsigned int* __restrict__ posO) {
  int i = blockIdx.x * blockDim.x + threadIdx.x;
  if (i < BS * A_) { cnt[i] = 0u; best[i] = 0ull; }
  if (i < BS * NGT) { posA[i] = 0u; posO[i] = 0u; }
}

// ------------------------------------------- K1: per-(b,g) top-13 + scatter
__global__ __launch_bounds__(TPB)
void topk_scatter_kernel(const float* __restrict__ pd_scores,
                         const float* __restrict__ pd_bboxes,
                         const float* __restrict__ anc,
                         const int*   __restrict__ gt_labels,
                         const float* __restrict__ gt_bboxes,
                         const float* __restrict__ mask_gt,
                         unsigned int* __restrict__ cnt,
                         unsigned long long* __restrict__ best) {
  const int bg = blockIdx.x;
  if (mask_gt[bg] == 0.0f) return;  // block-uniform: masked gts contribute nothing
  const int b = bg / NGT;
  const int g = bg % NGT;
  const int tid = threadIdx.x;

  const float gx0 = gt_bboxes[bg * 4 + 0];
  const float gy0 = gt_bboxes[bg * 4 + 1];
  const float gx1 = gt_bboxes[bg * 4 + 2];
  const float gy1 = gt_bboxes[bg * 4 + 3];
  const float garea = (gx1 - gx0) * (gy1 - gy0);
  const int lab = gt_labels[bg];

  float tv[TOPK_];
  int   ti[TOPK_];
#pragma unroll
  for (int k = 0; k < TOPK_; ++k) { tv[k] = -1.0f; ti[k] = 0x7fffffff; }

  const float* pb = pd_bboxes + (size_t)b * A_ * 4;
  const float* ps = pd_scores + (size_t)b * A_ * C_;

  for (int a = tid; a < A_; a += TPB) {
    if (a + TPB < A_) {
      __builtin_prefetch(&pb[(a + TPB) * 4], 0, 0);   // global_prefetch_b8
      __builtin_prefetch(&anc[(a + TPB) * 2], 0, 0);
    }
    const float ax = anc[a * 2 + 0];
    const float ay = anc[a * 2 + 1];
    float m = 0.0f;  // metric masked by in_gts, exactly as reference
    if ((ax - gx0 > EPSF) && (ay - gy0 > EPSF) && (gx1 - ax > EPSF) && (gy1 - ay > EPSF)) {
      const float px0 = pb[a * 4 + 0], py0 = pb[a * 4 + 1];
      const float px1 = pb[a * 4 + 2], py1 = pb[a * 4 + 3];
      const float iou = iou_box(gx0, gy0, gx1, gy1, garea, px0, py0, px1, py1);
      const float s = ps[(size_t)a * C_ + lab];
      const float i2 = iou * iou;
      m = s * i2 * i2 * i2;  // score^1 * iou^6
    }
    // sorted-descending insert, ties prefer lower index (matches lax.top_k)
    if (m > tv[TOPK_ - 1] || (m == tv[TOPK_ - 1] && a < ti[TOPK_ - 1])) {
      int k = TOPK_ - 1;
      while (k > 0 && (m > tv[k - 1] || (m == tv[k - 1] && a < ti[k - 1]))) {
        tv[k] = tv[k - 1]; ti[k] = ti[k - 1]; --k;
      }
      tv[k] = m; ti[k] = a;
    }
  }

  __shared__ float sv[TPB * TOPK_];
  __shared__ int   si[TPB * TOPK_];
#pragma unroll
  for (int k = 0; k < TOPK_; ++k) { sv[tid * TOPK_ + k] = tv[k]; si[tid * TOPK_ + k] = ti[k]; }
  __syncthreads();

  // pairwise tree merge of sorted 13-lists
  for (int step = 1; step < TPB; step <<= 1) {
    if ((tid & (2 * step - 1)) == 0) {
      float mv[TOPK_]; int mi[TOPK_];
      const float* Av = &sv[tid * TOPK_];           const int* Ai = &si[tid * TOPK_];
      const float* Bv = &sv[(tid + step) * TOPK_];  const int* Bi = &si[(tid + step) * TOPK_];
      int i = 0, j = 0;
      for (int k = 0; k < TOPK_; ++k) {
        bool takeA;
        if (j >= TOPK_)      takeA = true;
        else if (i >= TOPK_) takeA = false;
        else takeA = (Av[i] > Bv[j]) || (Av[i] == Bv[j] && Ai[i] <= Bi[j]);
        if (takeA) { mv[k] = Av[i]; mi[k] = Ai[i]; ++i; }
        else       { mv[k] = Bv[j]; mi[k] = Bi[j]; ++j; }
      }
      for (int k = 0; k < TOPK_; ++k) { sv[tid * TOPK_ + k] = mv[k]; si[tid * TOPK_ + k] = mi[k]; }
    }
    __syncthreads();
  }

  // scatter the 13 winners: candidate mask_pos entries (mask_topk * in_gts * mask_gt)
  if (tid < TOPK_) {
    const float v = sv[tid];
    const int   a = si[tid];
    if (a >= 0 && a < A_ && v >= 0.0f) {
      const float ax = anc[a * 2 + 0];
      const float ay = anc[a * 2 + 1];
      if ((ax - gx0 > EPSF) && (ay - gy0 > EPSF) && (gx1 - ax > EPSF) && (gy1 - ay > EPSF)) {
        const float px0 = pb[a * 4 + 0], py0 = pb[a * 4 + 1];
        const float px1 = pb[a * 4 + 2], py1 = pb[a * 4 + 3];
        const float iou = iou_box(gx0, gy0, gx1, gy1, garea, px0, py0, px1, py1);
        const unsigned long long packed =
            ((unsigned long long)__float_as_uint(iou) << 32) | (unsigned)g;
        atomicAdd(&cnt[b * A_ + a], 1u);
        atomicMax(&best[b * A_ + a], packed);  // iou >= 0 -> bit-monotonic
      }
    }
  }
}

// ------------------------- K2: per-(b,a) conflict resolution + gt normalizers
__global__ __launch_bounds__(256)
void resolve_kernel(const float* __restrict__ pd_scores,
                    const float* __restrict__ pd_bboxes,
                    const float* __restrict__ gt_bboxes,
                    const int*   __restrict__ gt_labels,
                    const unsigned int* __restrict__ cnt,
                    const unsigned long long* __restrict__ best,
                    int* __restrict__ gsel, float* __restrict__ fmet, float* __restrict__ fovl,
                    unsigned int* __restrict__ posA, unsigned int* __restrict__ posO) {
  const int idx = blockIdx.x * blockDim.x + threadIdx.x;
  if (idx >= BS * A_) return;
  const int b = idx / A_;
  const unsigned c = cnt[idx];
  if (c == 0u) { gsel[idx] = -1; fmet[idx] = 0.0f; fovl[idx] = 0.0f; return; }

  int g; float ovl;
  if (c == 1u) {
    const unsigned long long p = best[idx];
    g   = (int)(p & 0xffffffffull);
    ovl = __uint_as_float((unsigned)(p >> 32));
  } else {
    // reference: argmax of IoU over ALL gts (mask ignored), first-max tie-break
    const float px0 = pd_bboxes[(size_t)idx * 4 + 0];
    const float py0 = pd_bboxes[(size_t)idx * 4 + 1];
    const float px1 = pd_bboxes[(size_t)idx * 4 + 2];
    const float py1 = pd_bboxes[(size_t)idx * 4 + 3];
    g = 0; ovl = -1.0f;
    for (int gg = 0; gg < NGT; ++gg) {
      const float gx0 = gt_bboxes[(b * NGT + gg) * 4 + 0];
      const float gy0 = gt_bboxes[(b * NGT + gg) * 4 + 1];
      const float gx1 = gt_bboxes[(b * NGT + gg) * 4 + 2];
      const float gy1 = gt_bboxes[(b * NGT + gg) * 4 + 3];
      const float iou = iou_box(gx0, gy0, gx1, gy1, (gx1 - gx0) * (gy1 - gy0),
                                px0, py0, px1, py1);
      if (iou > ovl) { ovl = iou; g = gg; }
    }
  }
  const int lab = gt_labels[b * NGT + g];
  const float s  = pd_scores[(size_t)idx * C_ + lab];
  const float i2 = ovl * ovl;
  const float m  = s * i2 * i2 * i2;  // raw align metric at the final position
  gsel[idx] = g; fovl[idx] = ovl; fmet[idx] = m;
  atomicMax(&posA[b * NGT + g], __float_as_uint(m));
  atomicMax(&posO[b * NGT + g], __float_as_uint(ovl));
}

// ---------------- K3: labels / bboxes / fg_mask / per-anchor norm factor
__global__ __launch_bounds__(256)
void finalize_kernel(const int* __restrict__ gt_labels, const float* __restrict__ gt_bboxes,
                     const int* __restrict__ gsel, const float* __restrict__ fmet,
                     const unsigned int* __restrict__ posA, const unsigned int* __restrict__ posO,
                     float* __restrict__ nrm, float* __restrict__ out) {
  const int idx = blockIdx.x * blockDim.x + threadIdx.x;
  if (idx >= BS * A_) return;
  const int b = idx / A_;
  const int gs = gsel[idx];
  const int g0 = (gs >= 0) ? gs : 0;  // argmax of all-zero mask_pos is 0

  int lab = gt_labels[b * NGT + g0];
  if (lab < 0) lab = 0;
  out[idx] = (float)lab;  // target_labels (as float)

  float* ob = out + (size_t)BS * A_ + (size_t)idx * 4;  // target_bboxes
  ob[0] = gt_bboxes[(b * NGT + g0) * 4 + 0];
  ob[1] = gt_bboxes[(b * NGT + g0) * 4 + 1];
  ob[2] = gt_bboxes[(b * NGT + g0) * 4 + 2];
  ob[3] = gt_bboxes[(b * NGT + g0) * 4 + 3];

  out[(size_t)BS * A_ * 85 + idx] = (gs >= 0) ? 1.0f : 0.0f;  // fg_mask

  float nv = 0.0f;
  if (gs >= 0) {
    const float pa = __uint_as_float(posA[b * NGT + gs]);
    const float po = __uint_as_float(posO[b * NGT + gs]);
    nv = fmet[idx] * po / (pa + EPSF);
  }
  nrm[idx] = nv;
}

// ------- K4: target_scores = onehot(assign)^T . onehot(label) * norm via WMMA
// One wave per 16-anchor tile. K = 64 (60 gts padded) as two f16 16x16x32 WMMAs.
// {0,1} operands are exact in f16; f32 scaling applied at store.
__global__ __launch_bounds__(32)
void scores_wmma_kernel(const int* __restrict__ gsel, const float* __restrict__ nrm,
                        const int* __restrict__ gt_labels, float* __restrict__ out_scores) {
  const int tile = blockIdx.x;
  const int b  = tile / (A_ / 16);
  const int m0 = (tile % (A_ / 16)) * 16;
  const int lane = threadIdx.x;
  const int half = (lane < 16) ? 0 : 1;

  __shared__ int lab[64];
  for (int g = lane; g < 64; g += 32) lab[g] = (g < NGT) ? gt_labels[b * NGT + g] : -1;
  __syncthreads();

  // A fragment: A[m][k] = (gsel[m]==k), 16-bit A 16x32 lane layout (ISA 7.12.2)
  const int gme = gsel[b * A_ + m0 + (lane & 15)];
  v16h a0 = {}, a1 = {};
#pragma unroll
  for (int i = 0; i < 8; ++i) {
#pragma unroll
    for (int p = 0; p < 2; ++p) {
      const int e = 2 * i + p;
      const int k = (i < 4) ? ((half ? 8 : 0) + e) : ((half ? 24 : 16) + (e - 8));
      a0[e] = (gme == k)      ? (_Float16)1.0f : (_Float16)0.0f;  // g 0..31
      a1[e] = (gme == k + 32) ? (_Float16)1.0f : (_Float16)0.0f;  // g 32..63
    }
  }

  float nr[8];  // norm for the 8 D-rows this lane stores
#pragma unroll
  for (int r = 0; r < 8; ++r) nr[r] = nrm[b * A_ + m0 + r + (half ? 8 : 0)];

  for (int n0 = 0; n0 < C_; n0 += 16) {
    const int cclass = n0 + (lane & 15);
    // B fragment: B[k][n] = (label[k]==n); B 32x16 layout: K(e) = e<8 ? 8*half+e : 16+8*half+(e-8)
    v16h b0 = {}, b1 = {};
#pragma unroll
    for (int e = 0; e < 16; ++e) {
      const int k = (e < 8) ? ((half ? 8 : 0) + e) : ((half ? 24 : 16) + (e - 8));
      b0[e] = (lab[k]      == cclass) ? (_Float16)1.0f : (_Float16)0.0f;
      b1[e] = (lab[k + 32] == cclass) ? (_Float16)1.0f : (_Float16)0.0f;
    }
    v8f acc = {};
    acc = __builtin_amdgcn_wmma_f32_16x16x32_f16(false, a0, false, b0, (short)0, acc, false, false);
    acc = __builtin_amdgcn_wmma_f32_16x16x32_f16(false, a1, false, b1, (short)0, acc, false, false);
#pragma unroll
    for (int r = 0; r < 8; ++r) {
      const int row = m0 + r + (half ? 8 : 0);
      out_scores[(size_t)(b * A_ + row) * C_ + cclass] = acc[r] * nr[r];
    }
  }
}

// ---------------------------------------------------------------------------
extern "C" void kernel_launch(void* const* d_in, const int* in_sizes, int n_in,
                              void* d_out, int out_size, void* d_ws, size_t ws_size,
                              hipStream_t stream) {
  (void)in_sizes; (void)n_in; (void)out_size; (void)ws_size;
  const float* pd_scores = (const float*)d_in[0];
  const float* pd_bboxes = (const float*)d_in[1];
  const float* anc       = (const float*)d_in[2];
  const int*   gt_labels = (const int*)d_in[3];
  const float* gt_bboxes = (const float*)d_in[4];
  const float* mask_gt   = (const float*)d_in[5];
  float* out = (float*)d_out;

  char* p = (char*)d_ws;
  auto carve = [&](size_t bytes) -> char* {
    char* q = p;
    p += (bytes + 255) & ~(size_t)255;
    return q;
  };
  const size_t nBA = (size_t)BS * A_;
  unsigned long long* best = (unsigned long long*)carve(nBA * 8);
  unsigned int* cnt  = (unsigned int*)carve(nBA * 4);
  int*          gsel = (int*)carve(nBA * 4);
  float*        fmet = (float*)carve(nBA * 4);
  float*        fovl = (float*)carve(nBA * 4);
  float*        nrm  = (float*)carve(nBA * 4);
  unsigned int* posA = (unsigned int*)carve((size_t)BS * NGT * 4);
  unsigned int* posO = (unsigned int*)carve((size_t)BS * NGT * 4);

  const int blocksBA = (int)((nBA + 255) / 256);
  zero_kernel<<<blocksBA, 256, 0, stream>>>(cnt, best, posA, posO);
  topk_scatter_kernel<<<BS * NGT, TPB, 0, stream>>>(pd_scores, pd_bboxes, anc, gt_labels,
                                                    gt_bboxes, mask_gt, cnt, best);
  resolve_kernel<<<blocksBA, 256, 0, stream>>>(pd_scores, pd_bboxes, gt_bboxes, gt_labels,
                                               cnt, best, gsel, fmet, fovl, posA, posO);
  finalize_kernel<<<blocksBA, 256, 0, stream>>>(gt_labels, gt_bboxes, gsel, fmet,
                                                posA, posO, nrm, out);
  scores_wmma_kernel<<<BS * (A_ / 16), 32, 0, stream>>>(gsel, nrm, gt_labels,
                                                        out + nBA * 5);
}
